// MultiHeadAttention_71880572666214
// MI455X (gfx1250) — compile-verified, matrix operand order checked by reading, not run on hardware
//
#include <hip/hip_runtime.h>

typedef __attribute__((ext_vector_type(16))) _Float16 v16h;
typedef __attribute__((ext_vector_type(8)))  float    v8f;

#define B_   16
#define S_   7500
#define H_   64
#define T_   300     // frames per joint group
#define J_   25      // joint groups (S/T)
#define HS_  4       // heads
#define HD_  256     // hs*dk
#define TP_  320     // frames padded to 20 tiles of 16 (10 k-chunks of 32)
#define NT_  20      // key tiles (TP/16)
#define NQT_ 19      // query tiles (ceil(300/16))
#define NKC_ 10      // PV k-chunks (TP/32)

// ---- shared memory layout (bytes), total 144512 ----
// sK   @      0 : 320*64 f16  (K, row-major, padded rows zero)      40960
// sVT  @  40960 : 64*320 f16  (V transposed: [n][frame])            40960
// sWA  @  81920 : 64*64  f16  (phase1: Wk_h^T ; phase2: Wq_h^T)      8192
// sWB  @  90112 : 64*64  f16  (phase1: Wv_h^T ; phase2: Wo_h^T)      8192
// sP   @  98304 : phase1: 8 x (16x64) per-wave X staging (16 KB)
//                 phase2: 16x320 f16 probs (10240 B) + Xq stage @+12288
// sS   @ 114688 : 16*320 f32 score band                             20480
// sQ   @ 135168 : 16*64 f16                                          2048
// sO   @ 137216 : 16*64 f16                                          2048
// sRed @ 139264 : 16*16 f32 reduction scratch                        1024
// sMax @ 140288 : 16 f32 ; sInv @ 140352 : 16 f32
// sPar @ 140416 : 4 x 16x16 f32 split-k partials                     4096
#define SMEM_BYTES 144512

__device__ __forceinline__ v16h frag(const _Float16* base, int row0, int col0,
                                     int ld, int lane) {
  const _Float16* p = base + (row0 + (lane & 15)) * ld + col0 + ((lane >> 4) << 4);
  return *(const v16h*)p;
}

__device__ __forceinline__ v8f wmma_f16(v16h a, v16h b, v8f c) {
  return __builtin_amdgcn_wmma_f32_16x16x32_f16(false, a, false, b, (short)0, c,
                                                false, false);
}

// stage a 16x64 fp32 input tile (rows t*16.. of the 300-frame block) as f16;
// wave-private destination, rows >= 300 zero-padded
__device__ __forceinline__ void stage_x_wave(const float* __restrict__ xg,
                                             _Float16* dst, int b, int j, int t,
                                             int lane) {
  const size_t rowbase = ((size_t)b * S_ + (size_t)j * T_) * H_;
  for (int e = lane; e < 16 * 64; e += 32) {
    int rr = e >> 6, cc = e & 63;
    int s = t * 16 + rr;
    float v = (s < T_) ? xg[rowbase + (size_t)s * H_ + cc] : 0.0f;
    dst[rr * 64 + cc] = (_Float16)v;
  }
}

__global__ __launch_bounds__(256) void
mha_fused_kernel(const float* __restrict__ qIn, const float* __restrict__ kIn,
                 const float* __restrict__ vIn, const float* __restrict__ Wq,
                 const float* __restrict__ Wk, const float* __restrict__ Wv,
                 const float* __restrict__ Wo, float* __restrict__ out) {
  __shared__ __align__(128) unsigned char smem[SMEM_BYTES];
  _Float16* sK   = (_Float16*)(smem);
  _Float16* sVT  = (_Float16*)(smem + 40960);
  _Float16* sWA  = (_Float16*)(smem + 81920);
  _Float16* sWB  = (_Float16*)(smem + 90112);
  _Float16* sP   = (_Float16*)(smem + 98304);          // probs / wave staging
  _Float16* sXq  = (_Float16*)(smem + 98304 + 12288);  // 16x64 Xq stage
  float*    sS   = (float*)(smem + 114688);
  _Float16* sQ   = (_Float16*)(smem + 135168);
  _Float16* sO   = (_Float16*)(smem + 137216);
  float*    sRed = (float*)(smem + 139264);
  float*    sMax = (float*)(smem + 140288);
  float*    sInv = (float*)(smem + 140352);
  float*    sPar = (float*)(smem + 140416);

  const int tid  = threadIdx.x;
  const int lane = tid & 31;
  const int wave = tid >> 5;

  const int bid = blockIdx.x;         // 0..1599
  const int h   = bid & 3;
  const int j   = (bid >> 2) % J_;
  const int b   = bid / (J_ * HS_);

  // ---------- stage Wk_h^T, Wv_h^T (B-operand wants W^T row-major) ----------
  for (int e = tid; e < 64 * 64; e += 256) {
    int n = e >> 6, kk = e & 63;
    sWA[e] = (_Float16)Wk[kk * HD_ + h * 64 + n];
    sWB[e] = (_Float16)Wv[kk * HD_ + h * 64 + n];
  }
  __syncthreads();

  _Float16* sXw = sP + wave * 1024;  // 16x64 halves, wave-private staging

  // ---------- phase 1a: K projection -> sK (row-major, 320x64) ----------
  for (int rnd = 0; rnd < 3; ++rnd) {
    int t = wave + rnd * 8;
    if (t < NT_) stage_x_wave(kIn, sXw, b, j, t, lane);
    __syncthreads();
    if (t < NT_) {
      v16h a0 = frag(sXw, 0, 0, 64, lane);    // A fragments reused over nt
      v16h a1 = frag(sXw, 0, 32, 64, lane);
      for (int nt = 0; nt < 4; ++nt) {
        v8f acc = {};
        acc = wmma_f16(a0, frag(sWA, nt * 16, 0, 64, lane), acc);
        acc = wmma_f16(a1, frag(sWA, nt * 16, 32, 64, lane), acc);
        for (int r = 0; r < 8; ++r) {
          int m = r + ((lane >> 4) << 3);
          sK[(t * 16 + m) * 64 + nt * 16 + (lane & 15)] = (_Float16)acc[r];
        }
      }
    }
    __syncthreads();
  }

  // ---------- phase 1b: V projection, stored transposed -> sVT (64x320) ----
  for (int rnd = 0; rnd < 3; ++rnd) {
    int t = wave + rnd * 8;
    if (t < NT_) stage_x_wave(vIn, sXw, b, j, t, lane);
    __syncthreads();
    if (t < NT_) {
      v16h a0 = frag(sXw, 0, 0, 64, lane);
      v16h a1 = frag(sXw, 0, 32, 64, lane);
      for (int nt = 0; nt < 4; ++nt) {
        v8f acc = {};
        acc = wmma_f16(a0, frag(sWB, nt * 16, 0, 64, lane), acc);
        acc = wmma_f16(a1, frag(sWB, nt * 16, 32, 64, lane), acc);
        for (int r = 0; r < 8; ++r) {
          int m = r + ((lane >> 4) << 3);
          sVT[(nt * 16 + (lane & 15)) * TP_ + t * 16 + m] = (_Float16)acc[r];
        }
      }
    }
    __syncthreads();
  }

  // ---------- re-stage weights for phase 2: Wq_h^T, Wo_h^T ----------
  for (int e = tid; e < 64 * 64; e += 256) {
    int n = e >> 6, kk = e & 63;
    sWA[e] = (_Float16)Wq[kk * HD_ + h * 64 + n];
    sWB[e] = (_Float16)Wo[(h * 64 + kk) * H_ + n];
  }
  __syncthreads();

  const size_t rowbase = ((size_t)b * S_ + (size_t)j * T_) * H_;

  // ---------- phase 2: per 16-row query band ----------
  for (int qt = 0; qt < NQT_; ++qt) {
    // (1) stage Xq band cooperatively; prefetch the next band toward cache
    for (int e = tid; e < 16 * 64; e += 256) {
      int rr = e >> 6, cc = e & 63;
      int s = qt * 16 + rr;
      float v = (s < T_) ? qIn[rowbase + (size_t)s * H_ + cc] : 0.0f;
      sXq[e] = (_Float16)v;
    }
    if (qt + 1 < NQT_ && tid < 32) {       // 16 rows x 2 halves of 256B
      int rr = tid >> 1, cc = (tid & 1) * 32;
      int s = (qt + 1) * 16 + rr;
      if (s < T_) __builtin_prefetch(&qIn[rowbase + (size_t)s * H_ + cc], 0, 3);
    }
    __syncthreads();

    // (2) Q projection (scale 1/sqrt(dk) folded in), waves 0..3
    if (wave < 4) {
      int nt = wave;
      v16h a0 = frag(sXq, 0, 0, 64, lane);
      v16h a1 = frag(sXq, 0, 32, 64, lane);
      v8f acc = {};
      acc = wmma_f16(a0, frag(sWA, nt * 16, 0, 64, lane), acc);
      acc = wmma_f16(a1, frag(sWA, nt * 16, 32, 64, lane), acc);
      for (int r = 0; r < 8; ++r) {
        int m = r + ((lane >> 4) << 3);
        sQ[m * 64 + nt * 16 + (lane & 15)] = (_Float16)(acc[r] * 0.125f);
      }
    }
    __syncthreads();

    // (3) scores S = Q K^T over all 20 key tiles, spread across 8 waves;
    //     Q fragments hoisted out of the key-tile loop
    {
      v16h aq0 = frag(sQ, 0, 0, 64, lane);
      v16h aq1 = frag(sQ, 0, 32, 64, lane);
      for (int kt = wave; kt < NT_; kt += 8) {
        v8f acc = {};
        acc = wmma_f16(aq0, frag(sK, kt * 16, 0, 64, lane), acc);
        acc = wmma_f16(aq1, frag(sK, kt * 16, 32, 64, lane), acc);
        for (int r = 0; r < 8; ++r) {
          int m = r + ((lane >> 4) << 3);
          sS[m * TP_ + kt * 16 + (lane & 15)] = acc[r];
        }
      }
    }
    __syncthreads();

    // (4) softmax (unnormalized exp; 1/sum folded into PV epilogue)
    {
      int row = tid >> 4;  // 0..15
      int seg = tid & 15;  // 20 columns each
      float mx = -3.0e38f;
      for (int c = seg * 20; c < seg * 20 + 20; ++c)
        if (c < T_) mx = fmaxf(mx, sS[row * TP_ + c]);
      sRed[row * 16 + seg] = mx;
      __syncthreads();
      if (tid < 16) {
        float m2 = sRed[tid * 16];
        for (int i = 1; i < 16; ++i) m2 = fmaxf(m2, sRed[tid * 16 + i]);
        sMax[tid] = m2;
      }
      __syncthreads();
      float rm = sMax[row];
      float sum = 0.0f;
      for (int c = seg * 20; c < seg * 20 + 20; ++c) {
        float e = 0.0f;
        if (c < T_) { e = __expf(sS[row * TP_ + c] - rm); sum += e; }
        sP[row * TP_ + c] = (_Float16)e;
      }
      sRed[row * 16 + seg] = sum;
      __syncthreads();
      if (tid < 16) {
        float s2 = 0.0f;
        for (int i = 0; i < 16; ++i) s2 += sRed[tid * 16 + i];
        sInv[tid] = 1.0f / s2;
      }
    }
    __syncthreads();

    // (5) O = P V, split-k across all 8 waves: wave = grp*4 + nt,
    //     grp 0 -> chunks 0..4, grp 1 -> chunks 5..9; reduce via sPar
    {
      int nt  = wave & 3;
      int grp = wave >> 2;
      v8f acc = {};
      for (int kc = grp * 5; kc < grp * 5 + 5; ++kc)
        acc = wmma_f16(frag(sP, 0, kc * 32, TP_, lane),
                       frag(sVT, nt * 16, kc * 32, TP_, lane), acc);
      if (grp == 1) {
        for (int r = 0; r < 8; ++r) {
          int m = r + ((lane >> 4) << 3);
          sPar[nt * 256 + m * 16 + (lane & 15)] = acc[r];
        }
      }
      __syncthreads();
      if (grp == 0) {
        for (int r = 0; r < 8; ++r) {
          int m = r + ((lane >> 4) << 3);
          float o = (acc[r] + sPar[nt * 256 + m * 16 + (lane & 15)]) * sInv[m];
          sO[m * 64 + nt * 16 + (lane & 15)] = (_Float16)o;
        }
      }
    }
    __syncthreads();

    // (6) fold through Wo slice, split-k across 8 waves (1 WMMA each),
    //     then accumulate across heads into d_out with f32 global atomics
    {
      int nt = wave & 3;
      int kc = wave >> 2;
      v8f acc = {};
      acc = wmma_f16(frag(sO, 0, kc * 32, 64, lane),
                     frag(sWB, nt * 16, kc * 32, 64, lane), acc);
      if (kc == 1) {
        for (int r = 0; r < 8; ++r) {
          int m = r + ((lane >> 4) << 3);
          sPar[nt * 256 + m * 16 + (lane & 15)] = acc[r];
        }
      }
      __syncthreads();
      if (kc == 0) {
        for (int r = 0; r < 8; ++r) {
          int m = r + ((lane >> 4) << 3);
          int s = qt * 16 + m;
          if (s < T_) {
            float val = acc[r] + sPar[nt * 256 + m * 16 + (lane & 15)];
            unsafeAtomicAdd(&out[rowbase + (size_t)s * H_ + nt * 16 + (lane & 15)],
                            val);
          }
        }
      }
    }
    __syncthreads();
  }
}

extern "C" void kernel_launch(void* const* d_in, const int* in_sizes, int n_in,
                              void* d_out, int out_size, void* d_ws,
                              size_t ws_size, hipStream_t stream) {
  (void)in_sizes; (void)n_in; (void)d_ws; (void)ws_size;
  const float* q  = (const float*)d_in[0];
  const float* k  = (const float*)d_in[1];
  const float* v  = (const float*)d_in[2];
  const float* Wq = (const float*)d_in[3];
  const float* Wk = (const float*)d_in[4];
  const float* Wv = (const float*)d_in[5];
  const float* Wo = (const float*)d_in[6];
  float* out = (float*)d_out;

  // zero the output: heads accumulate via f32 global atomics
  hipMemsetAsync(d_out, 0, (size_t)out_size * sizeof(float), stream);

  dim3 grid(B_ * J_ * HS_);  // 1600 workgroups: one per (batch, group, head)
  mha_fused_kernel<<<grid, 256, 0, stream>>>(q, k, v, Wq, Wk, Wv, Wo, out);
}
